// MultiHeadAttention_2963527434617
// MI455X (gfx1250) — compile-verified, matrix operand order checked by reading, not run on hardware
//
#include <hip/hip_runtime.h>
#include <hip/hip_bf16.h>

// ---------------------------------------------------------------------------
// MultiHeadAttention (linear-attention variant) for MI455X / gfx1250.
//   * all GEMMs: bf16 WMMA (v_wmma_f32_16x16x32_bf16), fp32 accumulate
//   * tile staging: GLOBAL_LOAD_ASYNC_TO_LDS_B128 (ASYNCcnt) -> no staging
//     VGPRs, no conversion VALU in the K-loop, no spills
//   * fp32->bf16 conversion done once in a streaming pass (NT loads)
//   * heads-axis cumsum is row-local -> cheap elementwise pass
// ---------------------------------------------------------------------------

typedef __attribute__((ext_vector_type(16))) __bf16 v16bf;
typedef __attribute__((ext_vector_type(8)))  float  v8f;
typedef __attribute__((ext_vector_type(4)))  float  v4f;   // native vector for NT loads

#define D_MODEL 1024
#define M_ROWS  32768        // B*L = 4*8192
#define BLK_M   128
#define BLK_N   128
#define BLK_K   32
#define KSTEPS  (D_MODEL / BLK_K)   // 32
#define LDS_STRIDE 40        // halfwords per LDS row (padded, keeps 16B align)

__device__ __forceinline__ unsigned short f2bf(float f) {
    union { float f; unsigned int u; } v; v.f = f;
    unsigned int r = v.u + 0x7FFFu + ((v.u >> 16) & 1u);   // RNE
    return (unsigned short)(r >> 16);
}

__device__ __forceinline__ float bf2f(unsigned int lo16) {
    union { unsigned int u; float f; } v; v.u = lo16 << 16;
    return v.f;
}

// CDNA5 async copy: global -> LDS, 16 bytes per lane, tracked by ASYNCcnt.
__device__ __forceinline__ void async_b128(unsigned lds_addr, const void* gaddr) {
    asm volatile("global_load_async_to_lds_b128 %0, %1, off"
                 :: "v"(lds_addr), "v"(gaddr) : "memory");
}
__device__ __forceinline__ void wait_async0() {
    asm volatile("s_wait_asynccnt 0" ::: "memory");
}

// Load one 16x32 bf16 fragment (A layout; B tiles are pre-transposed so they
// use the identical layout). Two ds_load_b128 per lane:
//   lanes 0-15  : row = lane,     K base 0  -> VGPR0-3 = K0..7,  VGPR4-7 = K16..23
//   lanes 16-31 : row = lane-16,  K base 8  -> VGPR0-3 = K8..15, VGPR4-7 = K24..31
__device__ __forceinline__ v16bf frag_load(const unsigned short* lds, int row0, int lane) {
    const int r    = row0 + (lane & 15);
    const int koff = (lane >> 4) * 8;
    const unsigned short* p = lds + r * LDS_STRIDE + koff;
    union { uint4 q[2]; v16bf v; } u;
    u.q[0] = *(const uint4*)(p);
    u.q[1] = *(const uint4*)(p + 16);
    return u.v;
}

// ---------------------------------------------------------------------------
// Streaming fp32 -> bf16 convert (activations). NT loads: source not reused.
// ---------------------------------------------------------------------------
__global__ __launch_bounds__(256)
void cvt_kernel(const v4f* __restrict__ X, uint2* __restrict__ Y) {
    const int i = blockIdx.x * 256 + threadIdx.x;
    const v4f f = __builtin_nontemporal_load(X + i);
    uint2 p;
    p.x = (unsigned int)f2bf(f.x) | ((unsigned int)f2bf(f.y) << 16);
    p.y = (unsigned int)f2bf(f.z) | ((unsigned int)f2bf(f.w) << 16);
    Y[i] = p;
}

// ---------------------------------------------------------------------------
// Weight transpose + fp32->bf16 convert:  Wt[n][k] = bf16(W[k][n])
// ---------------------------------------------------------------------------
__global__ __launch_bounds__(256)
void wtrans_kernel(const float* __restrict__ W, unsigned short* __restrict__ Wt) {
    __shared__ float tile[32][33];
    const int k0 = blockIdx.y * 32;
    const int n0 = blockIdx.x * 32;
    const int tx = threadIdx.x & 31;
    const int ty = threadIdx.x >> 5;       // 0..7
#pragma unroll
    for (int i = 0; i < 32; i += 8)
        tile[ty + i][tx] = W[(size_t)(k0 + ty + i) * D_MODEL + n0 + tx];
    __syncthreads();
#pragma unroll
    for (int i = 0; i < 32; i += 8)
        Wt[(size_t)(n0 + ty + i) * D_MODEL + k0 + tx] = f2bf(tile[tx][ty + i]);
}

// ---------------------------------------------------------------------------
// Tiled bf16 WMMA GEMM:  Out = epilogue(A[M,K] @ Wt[N,K]^T + bias)
//   EPI=0 : bias only,    store bf16
//   EPI=1 : bias + elu+1, store bf16
//   EPI=2 : bias only,    store fp32
// Block = 256 threads = 8 waves, 128x128 tile; wave = 64x32 (4x2 WMMA tiles).
// Tiles staged via async global->LDS copies (double-buffered, 40 KB LDS).
// ---------------------------------------------------------------------------
template <int EPI>
__global__ __launch_bounds__(256, 1)
void gemm_kernel(const unsigned short* __restrict__ A,   // [M][K] bf16
                 const unsigned short* __restrict__ Bt,  // [N][K] bf16
                 const float* __restrict__ bias,
                 void* __restrict__ Out) {
    __shared__ __align__(16) unsigned short As[2][BLK_M * LDS_STRIDE];
    __shared__ __align__(16) unsigned short Bs[2][BLK_N * LDS_STRIDE];

    const int tid  = threadIdx.x;
    const int lane = tid & 31;
    const int wave = tid >> 5;
    const int wm   = wave & 1;          // 2 waves in M  -> 64 rows each
    const int wn   = wave >> 1;         // 4 waves in N  -> 32 cols each
    const int bm   = blockIdx.y * BLK_M;
    const int bn   = blockIdx.x * BLK_N;
    const int wmRow = wm * 64;
    const int wnRow = wn * 32;

    // Staging: 2 threads per tile row, 16 elements (32 B) each -> 2 async b128.
    const int ldRow = tid >> 1;             // 0..127
    const int ldSeg = (tid & 1) * 16;       // element offset within K-stage
    const unsigned short* gA = A  + (size_t)(bm + ldRow) * D_MODEL + ldSeg;
    const unsigned short* gB = Bt + (size_t)(bn + ldRow) * D_MODEL + ldSeg;
    const int ldsIdx = ldRow * LDS_STRIDE + ldSeg;

    auto stage = [&](int buf, int k0) {
        const unsigned aDst = (unsigned)(unsigned long long)&As[buf][ldsIdx];
        const unsigned bDst = (unsigned)(unsigned long long)&Bs[buf][ldsIdx];
        async_b128(aDst,      gA + k0);
        async_b128(aDst + 16, gA + k0 + 8);
        async_b128(bDst,      gB + k0);
        async_b128(bDst + 16, gB + k0 + 8);
    };

    v8f acc[4][2];
    const v8f vzero = {0.f, 0.f, 0.f, 0.f, 0.f, 0.f, 0.f, 0.f};
#pragma unroll
    for (int i = 0; i < 4; ++i)
#pragma unroll
        for (int j = 0; j < 2; ++j) acc[i][j] = vzero;

    // Prologue: fill buffer 0 and publish it.
    stage(0, 0);
    wait_async0();
    __syncthreads();

    for (int kt = 0; kt < KSTEPS; ++kt) {
        const int cur = kt & 1;
        if (kt + 1 < KSTEPS) stage(1 - cur, (kt + 1) * BLK_K);   // overlap with compute

        v16bf af[4], bf[2];
#pragma unroll
        for (int im = 0; im < 4; ++im) af[im] = frag_load(&As[cur][0], wmRow + im * 16, lane);
#pragma unroll
        for (int in = 0; in < 2; ++in) bf[in] = frag_load(&Bs[cur][0], wnRow + in * 16, lane);

#pragma unroll
        for (int im = 0; im < 4; ++im)
#pragma unroll
            for (int in = 0; in < 2; ++in)
                acc[im][in] = __builtin_amdgcn_wmma_f32_16x16x32_bf16(
                    false, af[im], false, bf[in], (short)0, acc[im][in], false, false);

        if (kt + 1 < KSTEPS) {
            wait_async0();       // our async writes into buffer (1-cur) done
            __syncthreads();     // everyone's writes visible, reads of cur done
        }
    }

    // Epilogue. C/D layout: VGPR r -> (M = r + 8*(lane>=16), N = lane&15).
    const int nIdx = lane & 15;
    const int mOff = (lane >> 4) * 8;
#pragma unroll
    for (int im = 0; im < 4; ++im) {
#pragma unroll
        for (int in = 0; in < 2; ++in) {
            const int gn  = bn + wnRow + in * 16 + nIdx;
            const int gm0 = bm + wmRow + im * 16 + mOff;
            const float bv = bias[gn];
#pragma unroll
            for (int r = 0; r < 8; ++r) {
                float x = acc[im][in][r] + bv;
                if constexpr (EPI == 1) x = (x > 0.f) ? (x + 1.f) : __expf(x);  // elu(x)+1
                const size_t idx = (size_t)(gm0 + r) * D_MODEL + gn;
                if constexpr (EPI == 2) ((float*)Out)[idx] = x;
                else                    ((unsigned short*)Out)[idx] = f2bf(x);
            }
        }
    }
}

// ---------------------------------------------------------------------------
// Heads-axis cumsum + elementwise:  attn = qp * (kp / cumsum_h(kp)) * vp
// Row-local: head h occupies columns [h*64, h*64+64). 32 lanes per row,
// each lane owns 2 consecutive depth positions, walks the 16 heads.
// ---------------------------------------------------------------------------
__global__ __launch_bounds__(256)
void cumsum_ew_kernel(const unsigned short* __restrict__ QP,
                      const unsigned short* __restrict__ KP,
                      const unsigned short* __restrict__ VP,
                      unsigned short* __restrict__ AT) {
    const int row = blockIdx.x * 8 + (threadIdx.x >> 5);
    const int dd  = (threadIdx.x & 31) * 2;
    const size_t base = (size_t)row * D_MODEL + dd;
    float c0 = 0.f, c1 = 0.f;
#pragma unroll
    for (int h = 0; h < 16; ++h) {
        const size_t off = base + h * 64;
        const unsigned int kp = *(const unsigned int*)(KP + off);
        const float k0 = bf2f(kp & 0xFFFFu), k1 = bf2f(kp >> 16);
        c0 += k0; c1 += k1;
        const unsigned int qp = *(const unsigned int*)(QP + off);
        const unsigned int vp = *(const unsigned int*)(VP + off);
        const float a0 = bf2f(qp & 0xFFFFu) * (k0 / c0) * bf2f(vp & 0xFFFFu);
        const float a1 = bf2f(qp >> 16)     * (k1 / c1) * bf2f(vp >> 16);
        *(unsigned int*)(AT + off) =
            (unsigned int)f2bf(a0) | ((unsigned int)f2bf(a1) << 16);
    }
}

// ---------------------------------------------------------------------------
// Launch graph:
//   weights -> bf16^T          (8 MB)
//   q/k/v   -> bf16            (192 MB)
//   3x projection GEMM -> bf16 (192 MB)
//   cumsum/elementwise -> AT   (aliases q-bf16 region; inputs dead by then)
//   final GEMM -> d_out (fp32)
// Scratch use: 392 MB of d_ws.
// ---------------------------------------------------------------------------
extern "C" void kernel_launch(void* const* d_in, const int* in_sizes, int n_in,
                              void* d_out, int out_size, void* d_ws, size_t ws_size,
                              hipStream_t stream) {
    (void)in_sizes; (void)n_in; (void)out_size; (void)ws_size;

    const float* v    = (const float*)d_in[0];
    const float* k    = (const float*)d_in[1];
    const float* q    = (const float*)d_in[2];
    const float* wq_w = (const float*)d_in[3];
    const float* wq_b = (const float*)d_in[4];
    const float* wk_w = (const float*)d_in[5];
    const float* wk_b = (const float*)d_in[6];
    const float* wv_w = (const float*)d_in[7];
    const float* wv_b = (const float*)d_in[8];
    const float* wc_w = (const float*)d_in[9];
    const float* wc_b = (const float*)d_in[10];

    const size_t WELE = (size_t)D_MODEL * D_MODEL;   // 1M elements
    const size_t PELE = (size_t)M_ROWS * D_MODEL;    // 32M elements

    unsigned short* base = (unsigned short*)d_ws;
    unsigned short* WtQ = base;
    unsigned short* WtK = WtQ + WELE;
    unsigned short* WtV = WtK + WELE;
    unsigned short* WtC = WtV + WELE;
    unsigned short* Qb  = WtC + WELE;     // bf16 activations
    unsigned short* Kb  = Qb + PELE;
    unsigned short* Vb  = Kb + PELE;
    unsigned short* QP  = Vb + PELE;      // bf16 projections
    unsigned short* KP  = QP + PELE;
    unsigned short* VP  = KP + PELE;
    unsigned short* AT  = Qb;             // reuse: Qb dead after projections

    const dim3 tgrid(32, 32);
    wtrans_kernel<<<tgrid, 256, 0, stream>>>(wq_w, WtQ);
    wtrans_kernel<<<tgrid, 256, 0, stream>>>(wk_w, WtK);
    wtrans_kernel<<<tgrid, 256, 0, stream>>>(wv_w, WtV);
    wtrans_kernel<<<tgrid, 256, 0, stream>>>(wc_w, WtC);

    const int cvtBlocks = (int)(PELE / 4 / 256);     // 32768
    cvt_kernel<<<cvtBlocks, 256, 0, stream>>>((const v4f*)q, (uint2*)Qb);
    cvt_kernel<<<cvtBlocks, 256, 0, stream>>>((const v4f*)k, (uint2*)Kb);
    cvt_kernel<<<cvtBlocks, 256, 0, stream>>>((const v4f*)v, (uint2*)Vb);

    const dim3 ggrid(D_MODEL / BLK_N, M_ROWS / BLK_M);   // (8, 256)
    gemm_kernel<1><<<ggrid, 256, 0, stream>>>(Qb, WtQ, wq_b, QP);  // elu+1
    gemm_kernel<1><<<ggrid, 256, 0, stream>>>(Kb, WtK, wk_b, KP);  // elu+1
    gemm_kernel<0><<<ggrid, 256, 0, stream>>>(Vb, WtV, wv_b, VP);  // plain

    cumsum_ew_kernel<<<M_ROWS / 8, 256, 0, stream>>>(QP, KP, VP, AT);

    gemm_kernel<2><<<ggrid, 256, 0, stream>>>(AT, WtC, wc_b, d_out);  // fp32 out
}
